// QuadraticEdgeUpdate_11854109737246
// MI455X (gfx1250) — matrix operand order repaired and checked by reading, not compiled, not run on hardware
//
#include <hip/hip_runtime.h>
#include <cstdint>

#define NDIM 384
#define DDIM 256
#define CDIM 128
#define MROWS (NDIM * NDIM)   // 147456

typedef __bf16 bf16;
typedef __attribute__((ext_vector_type(16))) __bf16 v16bf;
typedef __attribute__((ext_vector_type(8)))  __bf16 v8bf;
typedef __attribute__((ext_vector_type(4)))  __bf16 v4bf;
typedef __attribute__((ext_vector_type(8)))  float  v8f;

static __device__ __forceinline__ float sigmoidf_(float x) {
  return 1.0f / (1.0f + __expf(-x));
}

static __device__ __forceinline__ v16bf frag_join(v8bf lo, v8bf hi) {
  union { v16bf v; v8bf h[2]; } u;
  u.h[0] = lo; u.h[1] = hi;
  return u.v;
}

// A fragment: 16x32 tile (rows row0..+15, K k0..+31) from row-major bf16 (lda in elems).
// ISA layout: lanes 0-15 hold K 0-7 & 16-23 of row M=lane; lanes 16-31 hold K 8-15 & 24-31.
static __device__ __forceinline__ v16bf load_a_frag(const bf16* A, int lda,
                                                    int row0, int k0, int lane) {
  const int row = row0 + (lane & 15);
  const int kb  = k0 + ((lane >> 4) << 3);          // +8 for upper half-wave
  const bf16* p = A + (size_t)row * lda + kb;
  return frag_join(*(const v8bf*)p, *(const v8bf*)(p + 16));
}

// B fragment: 32x16 tile (K k0..+31, cols col0..+15) from N-major ("Bt") bf16 (ldb in elems).
// ISA layout: lanes 0-15 hold K 0-7 & 16-23 of col N=lane; lanes 16-31 hold K 8-15 & 24-31.
static __device__ __forceinline__ v16bf load_b_frag(const bf16* Bt, int ldb,
                                                    int col0, int k0, int lane) {
  const int col = col0 + (lane & 15);
  const int kb  = k0 + ((lane >> 4) << 3);          // +8 for upper half-wave
  const bf16* p = Bt + (size_t)col * ldb + kb;
  return frag_join(*(const v8bf*)p, *(const v8bf*)(p + 16));
}

#define WMMA_BF16(a, b, c) \
  __builtin_amdgcn_wmma_f32_16x16x32_bf16(false, (a), false, (b), (short)0, (c), false, false)

// Async global->LDS b128 copy; IOFFSET applies to both global and LDS sides (ISA 10.2).
#define ASYNC_LD32B(gp, ldsaddr)                                            \
  do {                                                                      \
    uint64_t ga_ = (uint64_t)(uintptr_t)(gp);                               \
    uint32_t la_ = (ldsaddr);                                               \
    asm volatile("global_load_async_to_lds_b128 %0, %1, off"                \
                 :: "v"(la_), "v"(ga_) : "memory");                         \
    asm volatile("global_load_async_to_lds_b128 %0, %1, off offset:16"      \
                 :: "v"(la_), "v"(ga_) : "memory");                         \
  } while (0)

// Flat pointer low 32 bits == LDS byte offset (ISA 10.2 LDS aperture truncation).
static __device__ __forceinline__ uint32_t lds_off(const void* p) {
  return (uint32_t)(uintptr_t)p;
}

// ---------------------------------------------------------------- K0: weight fp32 -> bf16
__global__ void __launch_bounds__(256)
k_f32_to_bf16(const float* __restrict__ s, bf16* __restrict__ d, int n) {
  int i = blockIdx.x * 256 + threadIdx.x;
  if (i < n) d[i] = (bf16)s[i];
}

// ---------------------------------------------------------------- K1: LayerNorm over D=256
__global__ void __launch_bounds__(256)
k_ln_in(const float* __restrict__ z, const float* __restrict__ w, const float* __restrict__ b,
        bf16* __restrict__ zn) {
  const int lane = threadIdx.x & 31;
  const int row  = blockIdx.x * 8 + (threadIdx.x >> 5);
  const float* x = z + (size_t)row * DDIM + lane * 8;
  float4 v0 = ((const float4*)x)[0];
  float4 v1 = ((const float4*)x)[1];
  float s = v0.x + v0.y + v0.z + v0.w + v1.x + v1.y + v1.z + v1.w;
#pragma unroll
  for (int off = 16; off > 0; off >>= 1) s += __shfl_xor(s, off, 32);
  const float mu = s * (1.0f / DDIM);
  float e[8] = {v0.x - mu, v0.y - mu, v0.z - mu, v0.w - mu,
                v1.x - mu, v1.y - mu, v1.z - mu, v1.w - mu};
  float vs = 0.0f;
#pragma unroll
  for (int t = 0; t < 8; ++t) vs += e[t] * e[t];
#pragma unroll
  for (int off = 16; off > 0; off >>= 1) vs += __shfl_xor(vs, off, 32);
  const float rstd = rsqrtf(vs * (1.0f / DDIM) + 1e-5f);
  v8bf o;
#pragma unroll
  for (int t = 0; t < 8; ++t) {
    const int c = lane * 8 + t;
    o[t] = (bf16)(e[t] * rstd * w[c] + b[c]);
  }
  *(v8bf*)(zn + (size_t)row * DDIM + lane * 8) = o;
}

// ---------------------------------------------------------------- K2: fused gated projection
// val = sigmoid(zn@G.T + gb) * (zn@W.T + b), channel-major [pair][o][m] bf16 out.
// Wave = 32x16 tile (2 M-tiles): W/G B-fragments reused across both -> 4 WMMA / 4 frag loads.
__global__ void __launch_bounds__(128)
k_gated_proj(const bf16* __restrict__ zn, const bf16* __restrict__ wbase,
             const float* __restrict__ ba, const float* __restrict__ gba,
             const float* __restrict__ bb, const float* __restrict__ gbb,
             const float* __restrict__ bc, const float* __restrict__ gbc,
             const float* __restrict__ bd, const float* __restrict__ gbd,
             bf16* __restrict__ outbase) {
  const int lane = threadIdx.x & 31;
  const int gw   = blockIdx.x * 4 + (threadIdx.x >> 5);
  const int mpairs = MROWS / 32;                    // 4608
  const int mp   = gw % mpairs;
  const int rest = gw / mpairs;                     // 0..31
  const int ctile = rest & 7;
  const int pair  = rest >> 3;                      // 0..3

  const bf16* W = wbase + (size_t)pair * 2 * CDIM * DDIM;
  const bf16* G = W + (size_t)CDIM * DDIM;
  const float* biasp; const float* gbiasp;
  switch (pair) {
    case 0:  biasp = ba; gbiasp = gba; break;
    case 1:  biasp = bb; gbiasp = gbb; break;
    case 2:  biasp = bc; gbiasp = gbc; break;
    default: biasp = bd; gbiasp = gbd; break;
  }

  v8f accw0 = {}, accw1 = {}, accg0 = {}, accg1 = {};
#pragma unroll
  for (int k0 = 0; k0 < DDIM; k0 += 32) {
    v16bf a0 = load_a_frag(zn, DDIM, mp * 32,      k0, lane);
    v16bf a1 = load_a_frag(zn, DDIM, mp * 32 + 16, k0, lane);
    v16bf wf = load_b_frag(W, DDIM, ctile * 16, k0, lane);
    v16bf gf = load_b_frag(G, DDIM, ctile * 16, k0, lane);
    accw0 = WMMA_BF16(a0, wf, accw0);
    accw1 = WMMA_BF16(a1, wf, accw1);
    accg0 = WMMA_BF16(a0, gf, accg0);
    accg1 = WMMA_BF16(a1, gf, accg1);
  }

  const int col  = ctile * 16 + (lane & 15);
  const float bv = biasp[col], gbv = gbiasp[col];
  bf16* colp = outbase + (size_t)pair * MROWS * CDIM + (size_t)col * MROWS;
  const int row0 = mp * 32 + ((lane >> 4) << 3);
  v8bf o0, o1;
#pragma unroll
  for (int r = 0; r < 8; ++r) {
    o0[r] = (bf16)((accw0[r] + bv) * sigmoidf_(accg0[r] + gbv));
    o1[r] = (bf16)((accw1[r] + bv) * sigmoidf_(accg1[r] + gbv));
  }
  *(v8bf*)(colp + row0)      = o0;                   // 8 rows M-contiguous
  *(v8bf*)(colp + row0 + 16) = o1;
}

// ---------------------------------------------------------------- K3: per-channel P.Q einsum
// k[i,j,o] = (A_o B_o^T)[i,j] * (C_o D_o^T)[i,j].
// Block = 128 thr = 4 waves, one 64x64 output tile per channel. K slabs (64x32 bf16 per
// operand) are staged into LDS with async global->LDS b128 copies, double-buffered on
// ASYNCcnt; each wave computes a 32x32 sub-tile as 2x2 WMMA accumulators (x2 for P and Q).
__global__ void __launch_bounds__(128)
k_pair_mul(const bf16* __restrict__ at, const bf16* __restrict__ bt,
           const bf16* __restrict__ ct, const bf16* __restrict__ dt,
           float* __restrict__ kout) {
  __shared__ __align__(16) bf16 smem[2][4][64 * 32];   // 2 stages x {A,B,C,D} x 4KB = 32KB
  const int tid  = threadIdx.x;
  const int lane = tid & 31;
  const int wave = tid >> 5;
  const int o    = blockIdx.x / 36;                    // (384/64)^2 = 36 tiles / channel
  const int rem  = blockIdx.x % 36;
  const int ib   = (rem / 6) * 64;
  const int jb   = (rem % 6) * 64;

  const bf16* pA = at + (size_t)o * MROWS;
  const bf16* pB = bt + (size_t)o * MROWS;
  const bf16* pC = ct + (size_t)o * MROWS;
  const bf16* pD = dt + (size_t)o * MROWS;

  const int srow  = tid >> 1;                          // 0..63: staged row per thread
  const int skoff = (tid & 1) * 16;                    // elem offset within 32-wide K slab

  auto issue_stage = [&](int ks, int buf) {
    const int kk = ks * 32 + skoff;
    ASYNC_LD32B(pA + (size_t)(ib + srow) * NDIM + kk, lds_off(&smem[buf][0][srow * 32 + skoff]));
    ASYNC_LD32B(pB + (size_t)(jb + srow) * NDIM + kk, lds_off(&smem[buf][1][srow * 32 + skoff]));
    ASYNC_LD32B(pC + (size_t)(ib + srow) * NDIM + kk, lds_off(&smem[buf][2][srow * 32 + skoff]));
    ASYNC_LD32B(pD + (size_t)(jb + srow) * NDIM + kk, lds_off(&smem[buf][3][srow * 32 + skoff]));
  };

  const int wi = (wave >> 1) * 32;                     // wave sub-tile within 64x64
  const int wj = (wave & 1) * 32;

  v8f p00 = {}, p01 = {}, p10 = {}, p11 = {};
  v8f q00 = {}, q01 = {}, q10 = {}, q11 = {};

  issue_stage(0, 0);
  for (int ks = 0; ks < NDIM / 32; ++ks) {             // 12 K steps
    const int buf = ks & 1;
    if (ks < NDIM / 32 - 1) {
      issue_stage(ks + 1, buf ^ 1);
      asm volatile("s_wait_asynccnt 0x8" ::: "memory"); // prev stage (in-order) complete
    } else {
      asm volatile("s_wait_asynccnt 0x0" ::: "memory");
    }
    __syncthreads();

    const bf16* As = &smem[buf][0][0];
    const bf16* Bs = &smem[buf][1][0];
    const bf16* Cs = &smem[buf][2][0];
    const bf16* Ds = &smem[buf][3][0];

    v16bf a0 = load_a_frag(As, 32, wi,      0, lane);
    v16bf a1 = load_a_frag(As, 32, wi + 16, 0, lane);
    v16bf b0 = load_b_frag(Bs, 32, wj,      0, lane);
    v16bf b1 = load_b_frag(Bs, 32, wj + 16, 0, lane);
    p00 = WMMA_BF16(a0, b0, p00);
    p01 = WMMA_BF16(a0, b1, p01);
    p10 = WMMA_BF16(a1, b0, p10);
    p11 = WMMA_BF16(a1, b1, p11);

    v16bf c0 = load_a_frag(Cs, 32, wi,      0, lane);
    v16bf c1 = load_a_frag(Cs, 32, wi + 16, 0, lane);
    v16bf d0 = load_b_frag(Ds, 32, wj,      0, lane);
    v16bf d1 = load_b_frag(Ds, 32, wj + 16, 0, lane);
    q00 = WMMA_BF16(c0, d0, q00);
    q01 = WMMA_BF16(c0, d1, q01);
    q10 = WMMA_BF16(c1, d0, q10);
    q11 = WMMA_BF16(c1, d1, q11);

    __syncthreads();                                   // protect buf^1 before next issue
  }

  const int jlo = jb + wj + (lane & 15);
  const int i0  = ib + wi + ((lane >> 4) << 3);
#pragma unroll
  for (int r = 0; r < 8; ++r) {
    const size_t ra = (size_t)(i0 + r) * NDIM;
    const size_t rb = (size_t)(i0 + 16 + r) * NDIM;
    kout[(ra + jlo)      * CDIM + o] = p00[r] * q00[r];
    kout[(ra + jlo + 16) * CDIM + o] = p01[r] * q01[r];
    kout[(rb + jlo)      * CDIM + o] = p10[r] * q10[r];
    kout[(rb + jlo + 16) * CDIM + o] = p11[r] * q11[r];
  }
}

// ---------------------------------------------------------------- K4: LayerNorm over C=128
__global__ void __launch_bounds__(256)
k_ln_out(const float* __restrict__ kb, const float* __restrict__ w, const float* __restrict__ b,
         bf16* __restrict__ kn) {
  const int lane = threadIdx.x & 31;
  const int row  = blockIdx.x * 8 + (threadIdx.x >> 5);
  const float* x = kb + (size_t)row * CDIM + lane * 4;
  float4 v = *(const float4*)x;
  float s = v.x + v.y + v.z + v.w;
#pragma unroll
  for (int off = 16; off > 0; off >>= 1) s += __shfl_xor(s, off, 32);
  const float mu = s * (1.0f / CDIM);
  float e[4] = {v.x - mu, v.y - mu, v.z - mu, v.w - mu};
  float vs = e[0]*e[0] + e[1]*e[1] + e[2]*e[2] + e[3]*e[3];
#pragma unroll
  for (int off = 16; off > 0; off >>= 1) vs += __shfl_xor(vs, off, 32);
  const float rstd = rsqrtf(vs * (1.0f / CDIM) + 1e-5f);
  v4bf o;
#pragma unroll
  for (int t = 0; t < 4; ++t) {
    const int c = lane * 4 + t;
    o[t] = (bf16)(e[t] * rstd * w[c] + b[c]);
  }
  *(v4bf*)(kn + (size_t)row * CDIM + lane * 4) = o;
}

// ---------------------------------------------------------------- K5: final gate * proj
// out = sigmoid(zn@Go.T + gbo) * (kn@Wo.T + bo), fp32 out (M x 256).
__global__ void __launch_bounds__(128)
k_final(const bf16* __restrict__ zn, const bf16* __restrict__ kn,
        const bf16* __restrict__ gob, const bf16* __restrict__ wob,
        const float* __restrict__ gbo, const float* __restrict__ bo,
        float* __restrict__ out) {
  const int lane = threadIdx.x & 31;
  const int gw   = blockIdx.x * 4 + (threadIdx.x >> 5);
  const int mtiles = MROWS / 16;                     // 9216
  const int mt = gw % mtiles;
  const int dt = gw / mtiles;                        // 0..15

  v8f accg = {}, accv = {};
#pragma unroll
  for (int k0 = 0; k0 < DDIM; k0 += 32) {            // gate GEMM, K=256
    v16bf af = load_a_frag(zn, DDIM, mt * 16, k0, lane);
    v16bf bf = load_b_frag(gob, DDIM, dt * 16, k0, lane);
    accg = WMMA_BF16(af, bf, accg);
  }
#pragma unroll
  for (int k0 = 0; k0 < CDIM; k0 += 32) {            // value GEMM, K=128
    v16bf af = load_a_frag(kn, CDIM, mt * 16, k0, lane);
    v16bf bf = load_b_frag(wob, CDIM, dt * 16, k0, lane);
    accv = WMMA_BF16(af, bf, accv);
  }

  const int d  = dt * 16 + (lane & 15);
  const int m0 = mt * 16 + ((lane >> 4) << 3);
  const float gbv = gbo[d], bv = bo[d];
#pragma unroll
  for (int r = 0; r < 8; ++r)
    out[(size_t)(m0 + r) * DDIM + d] = sigmoidf_(accg[r] + gbv) * (accv[r] + bv);
}

// ================================================================ host launcher
extern "C" void kernel_launch(void* const* d_in, const int* in_sizes, int n_in,
                              void* d_out, int out_size, void* d_ws, size_t ws_size,
                              hipStream_t stream) {
  (void)in_sizes; (void)n_in; (void)out_size; (void)ws_size;
  const float* z    = (const float*)d_in[0];
  const float* W_[4]  = {(const float*)d_in[1],  (const float*)d_in[5],
                         (const float*)d_in[9],  (const float*)d_in[13]};
  const float* bi_[4] = {(const float*)d_in[2],  (const float*)d_in[6],
                         (const float*)d_in[10], (const float*)d_in[14]};
  const float* G_[4]  = {(const float*)d_in[3],  (const float*)d_in[7],
                         (const float*)d_in[11], (const float*)d_in[15]};
  const float* gb_[4] = {(const float*)d_in[4],  (const float*)d_in[8],
                         (const float*)d_in[12], (const float*)d_in[16]};
  const float* ln_in_w = (const float*)d_in[17];
  const float* ln_in_b = (const float*)d_in[18];
  const float* ln_o_w  = (const float*)d_in[19];
  const float* ln_o_b  = (const float*)d_in[20];
  const float* Go  = (const float*)d_in[21];
  const float* gbo = (const float*)d_in[22];
  const float* Wo  = (const float*)d_in[23];
  const float* bo  = (const float*)d_in[24];
  float* out = (float*)d_out;

  const size_t M = MROWS;
  uint8_t* ws = (uint8_t*)d_ws;
  size_t off = 0;
  bf16*  zn   = (bf16*)(ws + off); off += M * DDIM * sizeof(bf16);
  bf16*  at   = (bf16*)(ws + off); off += M * CDIM * sizeof(bf16);
  bf16*  bt   = (bf16*)(ws + off); off += M * CDIM * sizeof(bf16);
  bf16*  ct   = (bf16*)(ws + off); off += M * CDIM * sizeof(bf16);
  bf16*  dt   = (bf16*)(ws + off); off += M * CDIM * sizeof(bf16);
  float* kbuf = (float*)(ws + off); off += M * CDIM * sizeof(float);
  bf16*  kn   = (bf16*)(ws + off); off += M * CDIM * sizeof(bf16);
  bf16*  wg   = (bf16*)(ws + off); off += 4 * 2 * (size_t)CDIM * DDIM * sizeof(bf16);
  bf16*  gob  = (bf16*)(ws + off); off += (size_t)DDIM * DDIM * sizeof(bf16);
  bf16*  wob  = (bf16*)(ws + off); off += (size_t)DDIM * CDIM * sizeof(bf16);

  // K0: convert weights to bf16 (already N-major for the WMMA B fragment)
  for (int p = 0; p < 4; ++p) {
    k_f32_to_bf16<<<(CDIM * DDIM + 255) / 256, 256, 0, stream>>>(
        W_[p], wg + (size_t)p * 2 * CDIM * DDIM, CDIM * DDIM);
    k_f32_to_bf16<<<(CDIM * DDIM + 255) / 256, 256, 0, stream>>>(
        G_[p], wg + (size_t)p * 2 * CDIM * DDIM + (size_t)CDIM * DDIM, CDIM * DDIM);
  }
  k_f32_to_bf16<<<(DDIM * DDIM + 255) / 256, 256, 0, stream>>>(Go, gob, DDIM * DDIM);
  k_f32_to_bf16<<<(DDIM * CDIM + 255) / 256, 256, 0, stream>>>(Wo, wob, DDIM * CDIM);

  // K1: input layernorm
  k_ln_in<<<M / 8, 256, 0, stream>>>(z, ln_in_w, ln_in_b, zn);

  // K2: gated projections a,b,c,d  (4608 m-pairs * 8 c-tiles * 4 pairs waves / 4 per block)
  k_gated_proj<<<(M / 32) * 8 * 4 / 4, 128, 0, stream>>>(
      zn, wg, bi_[0], gb_[0], bi_[1], gb_[1], bi_[2], gb_[2], bi_[3], gb_[3], at);

  // K3: per-channel batched matmul pair + product (128 ch * 36 blocks of 64x64)
  k_pair_mul<<<CDIM * 36, 128, 0, stream>>>(at, bt, ct, dt, kbuf);

  // K4: output layernorm over channels
  k_ln_out<<<M / 8, 256, 0, stream>>>(kbuf, ln_o_w, ln_o_b, kn);

  // K5: gated output projection (9216 m-tiles * 16 d-tiles / 4)
  k_final<<<(M / 16) * 16 / 4, 128, 0, stream>>>(zn, kn, gob, wob, gbo, bo, out);
}
